// CDAlignment_71373766525141
// MI455X (gfx1250) — matrix operand order repaired and checked by reading, not compiled, not run on hardware
//
#include <hip/hip_runtime.h>
#include <hip/hip_bf16.h>
#include <math.h>

// ---------------------------------------------------------------------------
// MI455X (gfx1250) implementation: all conv / deformable-conv GEMMs run on
// v_wmma_f32_16x16x32_bf16 (f32 accumulate). Activations live as NHWC bf16,
// offsets/masks as NHWC f32, weights pre-packed bf16 [Opad][kidx*Cin+ci].
// ---------------------------------------------------------------------------

typedef __bf16 bf16_t;
typedef __attribute__((ext_vector_type(16))) __bf16 v16bf;
typedef __attribute__((ext_vector_type(8)))  float  v8f;

union Frag16 { uint4 q[2]; v16bf v; };   // 32B fragment (16 bf16)
union H8     { uint4 q;    bf16_t h[8]; };

#define BB 4
#define HH 192
#define WW 192
#define XT (WW/16)              // 12 x-tiles per row
#define NBLK (BB*HH*XT)         // 9216 tile blocks
#define PIXELS ((size_t)BB*HH*WW)

// --------------------------- layout transforms -----------------------------

__global__ void nchw_to_nhwc_bf16(const float* __restrict__ in, bf16_t* __restrict__ out) {
    size_t total = PIXELS * 64;
    for (size_t t = blockIdx.x * (size_t)blockDim.x + threadIdx.x; t < total;
         t += (size_t)gridDim.x * blockDim.x) {
        size_t hw = t % ((size_t)HH * WW);
        int c = (int)((t / ((size_t)HH * WW)) % 64);
        int b = (int)(t / ((size_t)64 * HH * WW));
        out[((size_t)b * HH * WW + hw) * 64 + c] = (bf16_t)in[t];
    }
}

// w (O,Cin,3,3) f32  ->  bf16 [Opad][kidx*Cin + ci], zero padded rows
__global__ void pack_weights(const float* __restrict__ w, bf16_t* __restrict__ out,
                             int O, int Opad, int Cin) {
    int Kdim = Cin * 9;
    int total = Opad * Kdim;
    for (int t = blockIdx.x * blockDim.x + threadIdx.x; t < total;
         t += gridDim.x * blockDim.x) {
        int o = t / Kdim;
        int k = t % Kdim;
        int kidx = k / Cin;
        int ci   = k % Cin;
        float v = (o < O) ? w[(size_t)(o * Cin + ci) * 9 + kidx] : 0.0f;
        out[t] = (bf16_t)v;
    }
}

// ------------------------------ conv3x3 (WMMA) -----------------------------
// Block: 128 threads = 4 wave32. Tile: 16 pixels (one row segment) x CoutPad.
// Phase 1: im2col -> LDS (bf16, row stride Kdim+8 to dodge bank conflicts).
// Phase 2: waves do 16x16 WMMA tiles over K = 9*Cin in chunks of 32.
__global__ __launch_bounds__(128) void conv3x3_wmma(
    const bf16_t* __restrict__ inA,   // NHWC bf16, 64 ch
    const bf16_t* __restrict__ inB,   // second 64-ch buffer for concat (or null)
    int Cin,                          // 64 or 128
    const bf16_t* __restrict__ wpack, // [CoutPad][Kdim]
    const float*  __restrict__ bias,  // [Cout]
    int Cout, int CoutPad,
    bf16_t* __restrict__ out_bf,      // NHWC bf16 out (Cout==64) or null
    float*  __restrict__ out_f32,     // NHWC f32 out (stride Cout) or null
    int do_lrelu)
{
    __shared__ bf16_t sB[16 * (128 * 9 + 8)];   // max Cin=128 -> 37120 B
    const int Kdim = Cin * 9;
    const int rs   = Kdim + 8;

    int bp = blockIdx.x;
    int xt = bp % XT;
    int y  = (bp / XT) % HH;
    int b  = bp / (XT * HH);
    int x0 = xt * 16;

    // ---- phase 1: build im2col tile in LDS (8-channel chunks) ----
    int nc8 = Cin / 8;
    int ntask = 16 * 9 * nc8;
    for (int t = threadIdx.x; t < ntask; t += blockDim.x) {
        int i    = t & 15;
        int kidx = (t >> 4) % 9;
        int c8   = t / 144;
        int ky = kidx / 3, kx = kidx % 3;
        int yy = y + ky - 1;
        int xx = x0 + i + kx - 1;
        uint4 val = make_uint4(0u, 0u, 0u, 0u);
        if (yy >= 0 && yy < HH && xx >= 0 && xx < WW) {
            const bf16_t* src = (c8 < 8) ? inA : inB;
            int cl = (c8 & 7) * 8;
            size_t idx = ((size_t)(b * HH + yy) * WW + xx) * 64 + cl;
            val = *reinterpret_cast<const uint4*>(src + idx);
        }
        *reinterpret_cast<uint4*>(&sB[i * rs + kidx * Cin + c8 * 8]) = val;
    }
    __syncthreads();

    // ---- phase 2: WMMA GEMM ----
    int lane  = threadIdx.x & 31;
    int wv    = threadIdx.x >> 5;
    int lanem = lane & 15;
    int selA  = (lane < 16) ? 0 : 8;    // ISA A-frag K sub-block per lane half
    int selB  = (lane < 16) ? 0 : 8;    // ISA B-frag K sub-block per lane half
    int nKT = Kdim / 32;
    int nCT = CoutPad / 16;

    int px = x0 + lanem;
    size_t p = (size_t)(b * HH + y) * WW + px;
    int m0 = (lane < 16) ? 0 : 8;

    for (int ct = wv; ct < nCT; ct += 4) {
        int cb = ct * 16;
        v8f acc = {0.f, 0.f, 0.f, 0.f, 0.f, 0.f, 0.f, 0.f};
        const bf16_t* arow = wpack + (size_t)(cb + lanem) * Kdim;
        const bf16_t* brow = &sB[lanem * rs];
        for (int kt = 0; kt < nKT; ++kt) {
            int k0 = kt * 32;
            Frag16 af, bfr;
            af.q[0]  = *reinterpret_cast<const uint4*>(arow + k0 + selA);
            af.q[1]  = *reinterpret_cast<const uint4*>(arow + k0 + 16 + selA);
            bfr.q[0] = *reinterpret_cast<const uint4*>(brow + k0 + selB);
            bfr.q[1] = *reinterpret_cast<const uint4*>(brow + k0 + 16 + selB);
            acc = __builtin_amdgcn_wmma_f32_16x16x32_bf16(
                false, af.v, false, bfr.v, (short)0, acc, false, false);
        }
        if (out_bf) {  // Cout == 64 path, no predication needed
            H8 o;
            #pragma unroll
            for (int r = 0; r < 8; ++r) {
                int co = cb + m0 + r;
                float v = acc[r] + bias[co];
                if (do_lrelu) v = (v >= 0.f) ? v : 0.1f * v;
                o.h[r] = (bf16_t)v;
            }
            *reinterpret_cast<uint4*>(out_bf + p * 64 + cb + m0) = o.q;
        } else {       // f32 NHWC (offset conv, Cout=216 < CoutPad=224)
            #pragma unroll
            for (int r = 0; r < 8; ++r) {
                int co = cb + m0 + r;
                if (co < Cout) {
                    float v = acc[r] + bias[co];
                    if (do_lrelu) v = (v >= 0.f) ? v : 0.1f * v;
                    out_f32[p * (size_t)Cout + co] = v;
                }
            }
        }
    }
}

// ------------------- modulated deformable conv (WMMA) ----------------------
// x: NHWC bf16 (64ch). off: NHWC f32 (216ch): ch[d*18+2k]=dy, [d*18+2k+1]=dx,
// ch[144+d*9+k]=mask logits. Phase 1: bilinear-sample * sigmoid(mask) im2col
// into LDS. Phase 2: 64x16 WMMA GEMM over K=576.
__global__ __launch_bounds__(128) void dcn_wmma(
    const bf16_t* __restrict__ xin,
    const float*  __restrict__ off,
    const bf16_t* __restrict__ wpack,  // [64][576]
    const float*  __restrict__ bias,
    bf16_t* __restrict__ out_bf,       // NHWC bf16 or null
    float*  __restrict__ out_nchw,     // NCHW f32 (final) or null
    int do_lrelu)
{
    __shared__ bf16_t sB[16 * (576 + 8)];
    const int rs = 576 + 8;

    int bp = blockIdx.x;
    int xt = bp % XT;
    int y  = (bp / XT) % HH;
    int b  = bp / (XT * HH);
    int x0 = xt * 16;

    // ---- phase 1: sampled/masked im2col ----
    for (int t = threadIdx.x; t < 16 * 8 * 9; t += blockDim.x) {
        int i    = t & 15;       // pixel in tile
        int d    = (t >> 4) & 7; // deform group
        int kidx = t >> 7;       // tap 0..8
        int ky = kidx / 3, kx = kidx % 3;
        int xx = x0 + i;
        size_t p = (size_t)(b * HH + y) * WW + xx;
        const float* op = off + p * 216;
        float dy = op[d * 18 + 2 * kidx];
        float dx = op[d * 18 + 2 * kidx + 1];
        float mr = op[144 + d * 9 + kidx];
        float mask = 1.0f / (1.0f + expf(-mr));
        float py = (float)(y + ky - 1) + dy;
        float pxf = (float)(xx + kx - 1) + dx;
        float fy = floorf(py), fx = floorf(pxf);
        float wy = py - fy, wx = pxf - fx;
        int yi = (int)fy, xi = (int)fx;
        float acc[8] = {0, 0, 0, 0, 0, 0, 0, 0};
        #pragma unroll
        for (int cy = 0; cy < 2; ++cy) {
            int yc = yi + cy;
            if (yc < 0 || yc >= HH) continue;
            float wyc = cy ? wy : (1.f - wy);
            #pragma unroll
            for (int cx = 0; cx < 2; ++cx) {
                int xc = xi + cx;
                if (xc < 0 || xc >= WW) continue;
                float wgt = wyc * (cx ? wx : (1.f - wx));
                H8 v;
                v.q = *reinterpret_cast<const uint4*>(
                    xin + ((size_t)(b * HH + yc) * WW + xc) * 64 + d * 8);
                #pragma unroll
                for (int c = 0; c < 8; ++c) acc[c] += wgt * (float)v.h[c];
            }
        }
        H8 o;
        #pragma unroll
        for (int c = 0; c < 8; ++c) o.h[c] = (bf16_t)(acc[c] * mask);
        *reinterpret_cast<uint4*>(&sB[i * rs + kidx * 64 + d * 8]) = o.q;
    }
    __syncthreads();

    // ---- phase 2: GEMM, 4 waves x one 16-row co tile each ----
    int lane  = threadIdx.x & 31;
    int wv    = threadIdx.x >> 5;
    int lanem = lane & 15;
    int selA  = (lane < 16) ? 0 : 8;
    int selB  = (lane < 16) ? 0 : 8;
    int cb = wv * 16;

    v8f acc = {0.f, 0.f, 0.f, 0.f, 0.f, 0.f, 0.f, 0.f};
    const bf16_t* arow = wpack + (size_t)(cb + lanem) * 576;
    const bf16_t* brow = &sB[lanem * rs];
    for (int kt = 0; kt < 18; ++kt) {
        int k0 = kt * 32;
        Frag16 af, bfr;
        af.q[0]  = *reinterpret_cast<const uint4*>(arow + k0 + selA);
        af.q[1]  = *reinterpret_cast<const uint4*>(arow + k0 + 16 + selA);
        bfr.q[0] = *reinterpret_cast<const uint4*>(brow + k0 + selB);
        bfr.q[1] = *reinterpret_cast<const uint4*>(brow + k0 + 16 + selB);
        acc = __builtin_amdgcn_wmma_f32_16x16x32_bf16(
            false, af.v, false, bfr.v, (short)0, acc, false, false);
    }

    int px = x0 + lanem;
    size_t p = (size_t)(b * HH + y) * WW + px;
    int m0 = (lane < 16) ? 0 : 8;
    if (out_bf) {
        H8 o;
        #pragma unroll
        for (int r = 0; r < 8; ++r) {
            int co = cb + m0 + r;
            float v = acc[r] + bias[co];
            if (do_lrelu) v = (v >= 0.f) ? v : 0.1f * v;
            o.h[r] = (bf16_t)v;
        }
        *reinterpret_cast<uint4*>(out_bf + p * 64 + cb + m0) = o.q;
    } else {
        #pragma unroll
        for (int r = 0; r < 8; ++r) {
            int co = cb + m0 + r;
            float v = acc[r] + bias[co];
            if (do_lrelu) v = (v >= 0.f) ? v : 0.1f * v;
            out_nchw[((size_t)(b * 64 + co) * HH + y) * WW + px] = v;
        }
    }
}

// ------------------------------- launcher ----------------------------------

extern "C" void kernel_launch(void* const* d_in, const int* in_sizes, int n_in,
                              void* d_out, int out_size, void* d_ws, size_t ws_size,
                              hipStream_t stream) {
    const float* nbr      = (const float*)d_in[0];
    const float* ref      = (const float*)d_in[1];
    const float* oc1_w    = (const float*)d_in[2];
    const float* oc1_b    = (const float*)d_in[3];
    const float* oc2_w    = (const float*)d_in[4];
    const float* oc2_b    = (const float*)d_in[5];
    const float* d1off_w  = (const float*)d_in[6];
    const float* d1off_b  = (const float*)d_in[7];
    const float* dcn1_w   = (const float*)d_in[8];
    const float* dcn1_b   = (const float*)d_in[9];
    const float* fc_w     = (const float*)d_in[10];
    const float* fc_b     = (const float*)d_in[11];
    const float* cas1_w   = (const float*)d_in[12];
    const float* cas1_b   = (const float*)d_in[13];
    const float* cas2_w   = (const float*)d_in[14];
    const float* cas2_b   = (const float*)d_in[15];
    const float* cdoff_w  = (const float*)d_in[16];
    const float* cdoff_b  = (const float*)d_in[17];
    const float* casd_w   = (const float*)d_in[18];
    const float* casd_b   = (const float*)d_in[19];

    uint8_t* ws = (uint8_t*)d_ws;
    size_t cur = 0;
    auto alloc = [&](size_t bytes) -> void* {
        cur = (cur + 255) & ~(size_t)255;
        void* r = ws + cur;
        cur += bytes;
        return r;
    };

    const size_t SZ64 = PIXELS * 64 * sizeof(bf16_t);   // 18.9 MB
    bf16_t* b_nbr  = (bf16_t*)alloc(SZ64);
    bf16_t* b_ref  = (bf16_t*)alloc(SZ64);
    bf16_t* b_t1   = (bf16_t*)alloc(SZ64);
    bf16_t* b_t2   = (bf16_t*)alloc(SZ64);
    bf16_t* b_feat = (bf16_t*)alloc(SZ64);
    bf16_t* b_fc   = (bf16_t*)alloc(SZ64);
    float*  b_off  = (float*) alloc(PIXELS * 216 * sizeof(float));  // 127 MB

    bf16_t* wp_oc1   = (bf16_t*)alloc((size_t)64  * 1152 * 2);
    bf16_t* wp_oc2   = (bf16_t*)alloc((size_t)64  * 576  * 2);
    bf16_t* wp_d1off = (bf16_t*)alloc((size_t)224 * 576  * 2);
    bf16_t* wp_dcn1  = (bf16_t*)alloc((size_t)64  * 576  * 2);
    bf16_t* wp_fc    = (bf16_t*)alloc((size_t)64  * 576  * 2);
    bf16_t* wp_cas1  = (bf16_t*)alloc((size_t)64  * 1152 * 2);
    bf16_t* wp_cas2  = (bf16_t*)alloc((size_t)64  * 576  * 2);
    bf16_t* wp_cdoff = (bf16_t*)alloc((size_t)224 * 576  * 2);
    bf16_t* wp_casd  = (bf16_t*)alloc((size_t)64  * 576  * 2);

    float* out = (float*)d_out;

    // layout transforms
    nchw_to_nhwc_bf16<<<8192, 256, 0, stream>>>(nbr, b_nbr);
    nchw_to_nhwc_bf16<<<8192, 256, 0, stream>>>(ref, b_ref);
    pack_weights<<<256, 256, 0, stream>>>(oc1_w,   wp_oc1,   64,  64,  128);
    pack_weights<<<256, 256, 0, stream>>>(oc2_w,   wp_oc2,   64,  64,  64);
    pack_weights<<<256, 256, 0, stream>>>(d1off_w, wp_d1off, 216, 224, 64);
    pack_weights<<<256, 256, 0, stream>>>(dcn1_w,  wp_dcn1,  64,  64,  64);
    pack_weights<<<256, 256, 0, stream>>>(fc_w,    wp_fc,    64,  64,  64);
    pack_weights<<<256, 256, 0, stream>>>(cas1_w,  wp_cas1,  64,  64,  128);
    pack_weights<<<256, 256, 0, stream>>>(cas2_w,  wp_cas2,  64,  64,  64);
    pack_weights<<<256, 256, 0, stream>>>(cdoff_w, wp_cdoff, 216, 224, 64);
    pack_weights<<<256, 256, 0, stream>>>(casd_w,  wp_casd,  64,  64,  64);

    // offset = lrelu(conv(concat(nbr,ref), oc1)); lrelu(conv(., oc2))
    conv3x3_wmma<<<NBLK, 128, 0, stream>>>(b_nbr, b_ref, 128, wp_oc1, oc1_b,
                                           64, 64, b_t1, nullptr, 1);
    conv3x3_wmma<<<NBLK, 128, 0, stream>>>(b_t1, nullptr, 64, wp_oc2, oc2_b,
                                           64, 64, b_t2, nullptr, 1);
    // dcn1: offset conv (216ch, f32 NHWC), then modulated deform conv
    conv3x3_wmma<<<NBLK, 128, 0, stream>>>(b_t2, nullptr, 64, wp_d1off, d1off_b,
                                           216, 224, nullptr, b_off, 0);
    dcn_wmma<<<NBLK, 128, 0, stream>>>(b_nbr, b_off, wp_dcn1, dcn1_b,
                                       b_feat, nullptr, 0);
    // feat = conv(feat, fc)  (no activation)
    conv3x3_wmma<<<NBLK, 128, 0, stream>>>(b_feat, nullptr, 64, wp_fc, fc_b,
                                           64, 64, b_fc, nullptr, 0);
    // cascade offset branch: concat(feat, ref) -> cas1 -> cas2 (lrelu both)
    conv3x3_wmma<<<NBLK, 128, 0, stream>>>(b_fc, b_ref, 128, wp_cas1, cas1_b,
                                           64, 64, b_t1, nullptr, 1);
    conv3x3_wmma<<<NBLK, 128, 0, stream>>>(b_t1, nullptr, 64, wp_cas2, cas2_b,
                                           64, 64, b_t2, nullptr, 1);
    // cascade dcn: offset conv + deform conv, lrelu, f32 NCHW -> d_out
    conv3x3_wmma<<<NBLK, 128, 0, stream>>>(b_t2, nullptr, 64, wp_cdoff, cdoff_b,
                                           216, 224, nullptr, b_off, 0);
    dcn_wmma<<<NBLK, 128, 0, stream>>>(b_fc, b_off, wp_casd, casd_b,
                                       nullptr, out, 1);
}
